// SynthesizerCosineSimilarity_73933567034178
// MI455X (gfx1250) — matrix operand order repaired and checked by reading, not compiled, not run on hardware
//
#include <hip/hip_runtime.h>
#include <hip/hip_bf16.h>

// ---------------------------------------------------------------------------
// SynthesizerCosineSimilarity for MI455X (gfx1250, wave32, WMMA)
//
// B=4, N=4096, D=256, TOPK=64.
// Softmax identity: attn@value = (sum_topk (e^v-1)*value[idx] + colsum(value))
//                                / (sum_topk (e^v-1) + N)
// so the second N*N GEMM collapses to a 64-row gather. The N*N cosine GEMM
// runs on v_wmma_f32_16x16x32_f16; a 16-row x 4096-col slab (256KB) lives
// entirely in the WGP's 320KB LDS for in-place top-k selection. The shared
// 16-row A panel is staged into LDS with global_load_async_to_lds_b128.
// ---------------------------------------------------------------------------

#define B_    4
#define N_    4096
#define D_    256
#define TOPK_ 64
#define NSPLIT_ 64   // colsum split factor over N

typedef __attribute__((ext_vector_type(16))) _Float16 v16h;
typedef __attribute__((ext_vector_type(8)))  _Float16 v8h;
typedef __attribute__((ext_vector_type(4)))  _Float16 v4h;
typedef __attribute__((ext_vector_type(8)))  float    v8f;

// LDS layout for the fused kernel (bytes)
#define SIMS_BYTES   ((size_t)16 * N_ * 4)                  // 256 KB
#define TIDX_OFF     (SIMS_BYTES)
#define TW_OFF       (TIDX_OFF + (size_t)16 * TOPK_ * 4)
#define TS_OFF       (TW_OFF + (size_t)16 * TOPK_ * 4)
#define APANEL_OFF   (TS_OFF + 64)                          // 16B aligned
#define SMEM_BYTES   (APANEL_OFF + (size_t)16 * D_ * 2)     // + 8 KB A panel

// ---------------------------------------------------------------------------
// Kernel 1: row L2-normalize x -> nx (f16), and emit plain f16 copy xh.
// One wave32 per row of 256 floats.
// ---------------------------------------------------------------------------
__global__ __launch_bounds__(256) void prep_normalize(
    const float* __restrict__ x, _Float16* __restrict__ nx,
    _Float16* __restrict__ xh) {
  const int lane = threadIdx.x & 31;
  const int wave = threadIdx.x >> 5;
  const int row  = blockIdx.x * 8 + wave;          // over B*N rows
  const float* xr = x + (size_t)row * D_;
  float4 a = ((const float4*)xr)[lane * 2 + 0];
  float4 c = ((const float4*)xr)[lane * 2 + 1];
  float ss = a.x*a.x + a.y*a.y + a.z*a.z + a.w*a.w
           + c.x*c.x + c.y*c.y + c.z*c.z + c.w*c.w;
  #pragma unroll
  for (int off = 16; off > 0; off >>= 1) ss += __shfl_xor(ss, off, 32);
  const float s = 1.0f / fmaxf(sqrtf(ss), 1e-12f);
  v8h h;
  h[0] = (_Float16)(a.x * s); h[1] = (_Float16)(a.y * s);
  h[2] = (_Float16)(a.z * s); h[3] = (_Float16)(a.w * s);
  h[4] = (_Float16)(c.x * s); h[5] = (_Float16)(c.y * s);
  h[6] = (_Float16)(c.z * s); h[7] = (_Float16)(c.w * s);
  ((v8h*)(nx + (size_t)row * D_))[lane] = h;
  v8h p;
  p[0] = (_Float16)a.x; p[1] = (_Float16)a.y;
  p[2] = (_Float16)a.z; p[3] = (_Float16)a.w;
  p[4] = (_Float16)c.x; p[5] = (_Float16)c.y;
  p[6] = (_Float16)c.z; p[7] = (_Float16)c.w;
  ((v8h*)(xh + (size_t)row * D_))[lane] = p;
}

// ---------------------------------------------------------------------------
// Kernel 2: W (f32, DxD) -> Wh (f16). 64 blocks x 256 thr x 4 elems.
// ---------------------------------------------------------------------------
__global__ __launch_bounds__(256) void convert_w(
    const float* __restrict__ W, _Float16* __restrict__ Wh) {
  const int t = blockIdx.x * 256 + threadIdx.x;
  const float4 f = ((const float4*)W)[t];
  v4h h;
  h[0] = (_Float16)f.x; h[1] = (_Float16)f.y;
  h[2] = (_Float16)f.z; h[3] = (_Float16)f.w;
  ((v4h*)Wh)[t] = h;
}

// ---------------------------------------------------------------------------
// Kernel 3: value = x @ W^T + bias via v_wmma_f32_16x16x32_f16, pure-f16 loads.
// Block = 8 waves; each wave owns 2 of 16 column tiles of a 16-row panel.
// A layout per ISA 7.12.2 (16-bit A 16x32): lane<16 holds K {0..7,16..23},
// lane>=16 holds K {8..15,24..31} of row M=lane%16. B lanes hold the same
// K split of column e (= row e of W).
// ---------------------------------------------------------------------------
__global__ __launch_bounds__(256) void value_gemm(
    const _Float16* __restrict__ xh, const _Float16* __restrict__ Wh,
    const float* __restrict__ bias, float* __restrict__ value) {
  const int lane   = threadIdx.x & 31;
  const int wave   = threadIdx.x >> 5;
  const int lcol   = lane & 15;
  const int hi     = lane >> 4;                    // 0: lanes 0-15, 1: 16-31
  const int rowBase = blockIdx.x * 16;
  const _Float16* arow = xh + (size_t)(rowBase + lcol) * D_;

  #pragma unroll
  for (int t = 0; t < 2; ++t) {
    const int e = (wave + t * 8) * 16 + lcol;      // output / W row index
    const _Float16* brow = Wh + (size_t)e * D_;
    v8f acc = {};
    #pragma unroll
    for (int ks = 0; ks < 8; ++ks) {
      const int kbA = ks * 32 + hi * 8;
      v8h a0 = *(const v8h*)(arow + kbA);
      v8h a1 = *(const v8h*)(arow + kbA + 16);
      const int kbB = ks * 32 + hi * 8;
      v8h b0 = *(const v8h*)(brow + kbB);
      v8h b1 = *(const v8h*)(brow + kbB + 16);
      v16h A, Bv;
      #pragma unroll
      for (int j = 0; j < 8; ++j) {
        A[j]  = a0[j];  A[8 + j]  = a1[j];
        Bv[j] = b0[j];  Bv[8 + j] = b1[j];
      }
      acc = __builtin_amdgcn_wmma_f32_16x16x32_f16(
          false, A, false, Bv, (short)0, acc, false, false);
    }
    const float be = bias[e];
    #pragma unroll
    for (int r = 0; r < 8; ++r) {
      const int row = rowBase + r + hi * 8;        // C/D layout, ISA 7.12.2
      value[(size_t)row * D_ + e] = acc[r] + be;
    }
  }
}

// ---------------------------------------------------------------------------
// Kernel 4a/4b: deterministic two-pass colsum[b][d] = sum_n value[b][n][d]
// ---------------------------------------------------------------------------
__global__ __launch_bounds__(256) void colsum_partial(
    const float* __restrict__ value, float* __restrict__ partial) {
  const int c = blockIdx.x / 4;                    // n-chunk 0..NSPLIT_-1
  const int t = (blockIdx.x % 4) * 256 + threadIdx.x;  // 0..B*D-1
  const int b = t / D_, d = t % D_;
  const int n0 = c * (N_ / NSPLIT_);
  const float* p = value + (size_t)b * N_ * D_ + (size_t)n0 * D_ + d;
  float s = 0.0f;
  for (int n = 0; n < N_ / NSPLIT_; ++n) s += p[(size_t)n * D_];
  partial[(size_t)c * (B_ * D_) + t] = s;
}

__global__ __launch_bounds__(256) void colsum_reduce(
    const float* __restrict__ partial, float* __restrict__ colsum) {
  const int t = blockIdx.x * 256 + threadIdx.x;    // 0..B*D-1
  float s = 0.0f;
  for (int c = 0; c < NSPLIT_; ++c) s += partial[(size_t)c * (B_ * D_) + t];
  colsum[t] = s;
}

// ---------------------------------------------------------------------------
// Kernel 5: fused cos-GEMM (WMMA) -> LDS slab -> top-64 -> weighted gather
// 16 waves (512 thr). LDS: sims[16][4096] f32 + topk idx/w + 8KB A panel.
// ---------------------------------------------------------------------------
__global__ __launch_bounds__(512, 1) void cos_topk_out(
    const _Float16* __restrict__ nx, const float* __restrict__ value,
    const float* __restrict__ colsum, float* __restrict__ out) {
  extern __shared__ char smem[];
  float*    sims   = (float*)smem;                       // [16][N_]
  int*      tIdx   = (int*)  (smem + TIDX_OFF);          // [16][TOPK_]
  float*    tW     = (float*)(smem + TW_OFF);            // [16][TOPK_]
  float*    tS     = (float*)(smem + TS_OFF);            // [16]
  _Float16* aPanel = (_Float16*)(smem + APANEL_OFF);     // [16][D_]

  const int lane = threadIdx.x & 31;
  const int wave = threadIdx.x >> 5;               // 0..15
  const int lcol = lane & 15;
  const int hi   = lane >> 4;
  const int b       = blockIdx.x / (N_ / 16);
  const int rowBase = (blockIdx.x % (N_ / 16)) * 16;
  const _Float16* nb = nx + (size_t)b * N_ * D_;

  // ---- Stage the shared 16-row A panel (8KB) into LDS asynchronously:
  // 512 threads x one global_load_async_to_lds_b128 (16B) each.
  {
    const uint32_t o = (uint32_t)threadIdx.x * 16u;
    const uint32_t ldsOff = (uint32_t)(uintptr_t)aPanel + o;
    const uint64_t gaddr =
        (uint64_t)(uintptr_t)((const char*)(nb + (size_t)rowBase * D_) + o);
    asm volatile("global_load_async_to_lds_b128 %0, %1, off"
                 :: "v"(ldsOff), "v"(gaddr) : "memory");
    asm volatile("s_wait_asynccnt 0" ::: "memory");
  }
  __syncthreads();

  // ---- Phase 1: sims[16][4096] = nx[rows] @ nx^T (Gram: both operands are
  // row-major reads of nx). A panel loaded once per wave from LDS (64 VGPRs).
  v16h Areg[8];
  {
    const _Float16* arow = aPanel + lcol * D_;
    #pragma unroll
    for (int ks = 0; ks < 8; ++ks) {
      const int kb = ks * 32 + hi * 8;
      v8h lo = *(const v8h*)(arow + kb);           // ds_load_b128
      v8h hh = *(const v8h*)(arow + kb + 16);
      v16h A;
      #pragma unroll
      for (int j = 0; j < 8; ++j) { A[j] = lo[j]; A[8 + j] = hh[j]; }
      Areg[ks] = A;
    }
  }
  for (int nt = wave; nt < N_ / 16; nt += 16) {
    const _Float16* brow = nb + (size_t)(nt * 16 + lcol) * D_;
    v8f acc = {};
    #pragma unroll
    for (int ks = 0; ks < 8; ++ks) {
      const int kb = ks * 32 + hi * 8;
      v8h b0 = *(const v8h*)(brow + kb);
      v8h b1 = *(const v8h*)(brow + kb + 16);
      v16h Bv;
      #pragma unroll
      for (int j = 0; j < 8; ++j) { Bv[j] = b0[j]; Bv[8 + j] = b1[j]; }
      acc = __builtin_amdgcn_wmma_f32_16x16x32_f16(
          false, Areg[ks], false, Bv, (short)0, acc, false, false);
    }
    #pragma unroll
    for (int r = 0; r < 8; ++r)
      sims[(size_t)(r + hi * 8) * N_ + nt * 16 + lcol] = acc[r];
  }
  __syncthreads();

  // ---- Phase 2: top-64 per row, wave w owns row w (iterative argmax in LDS)
  {
    float* row = sims + (size_t)wave * N_;
    float ssum = 0.0f;
    for (int k = 0; k < TOPK_; ++k) {
      float mv = -3.0e38f; int mi = 0;
      for (int i = lane; i < N_; i += 32) {
        const float v = row[i];
        if (v > mv) { mv = v; mi = i; }
      }
      #pragma unroll
      for (int off = 16; off > 0; off >>= 1) {
        const float ov = __shfl_xor(mv, off, 32);
        const int   oi = __shfl_xor(mi, off, 32);
        if (ov > mv || (ov == mv && oi < mi)) { mv = ov; mi = oi; }
      }
      const float w = expm1f(mv);                  // e^v - 1
      ssum += w;
      if (lane == 0) {
        tIdx[wave * TOPK_ + k] = mi;
        tW  [wave * TOPK_ + k] = w;
        row[mi] = -3.0e38f;                        // knock out for next pass
      }
      // LDS RAW: lane0's knockout store must land before the next scan.
      asm volatile("s_wait_dscnt 0" ::: "memory");
      __builtin_amdgcn_wave_barrier();
    }
    if (lane == 0) tS[wave] = ssum;
  }
  __syncthreads();

  // ---- Phase 3: out[row] = (sum_k w_k*value[m_k] + colsum) / (S + N)
  {
    const float* vb = value + (size_t)b * N_ * D_;
    const int d0 = lane * 8;                       // 32 lanes x 8 = 256 = D
    float4 acc0 = *(const float4*)(colsum + b * D_ + d0);
    float4 acc1 = *(const float4*)(colsum + b * D_ + d0 + 4);
    const float rden = 1.0f / (tS[wave] + (float)N_);
    for (int k = 0; k < TOPK_; ++k) {
      const int   m = tIdx[wave * TOPK_ + k];
      const float w = tW  [wave * TOPK_ + k];
      if (k + 1 < TOPK_)
        __builtin_prefetch(vb + (size_t)tIdx[wave * TOPK_ + k + 1] * D_ + d0, 0, 0);
      const float* vr = vb + (size_t)m * D_ + d0;
      const float4 v0 = *(const float4*)(vr);
      const float4 v1 = *(const float4*)(vr + 4);
      acc0.x += w * v0.x; acc0.y += w * v0.y; acc0.z += w * v0.z; acc0.w += w * v0.w;
      acc1.x += w * v1.x; acc1.y += w * v1.y; acc1.z += w * v1.z; acc1.w += w * v1.w;
    }
    acc0.x *= rden; acc0.y *= rden; acc0.z *= rden; acc0.w *= rden;
    acc1.x *= rden; acc1.y *= rden; acc1.z *= rden; acc1.w *= rden;
    float* orow = out + ((size_t)b * N_ + rowBase + wave) * D_ + d0;
    *(float4*)(orow)     = acc0;
    *(float4*)(orow + 4) = acc1;
  }
}

// ---------------------------------------------------------------------------
extern "C" void kernel_launch(void* const* d_in, const int* in_sizes, int n_in,
                              void* d_out, int out_size, void* d_ws, size_t ws_size,
                              hipStream_t stream) {
  (void)in_sizes; (void)n_in; (void)out_size; (void)ws_size;
  const float* x    = (const float*)d_in[0];
  const float* W    = (const float*)d_in[1];
  const float* bias = (const float*)d_in[2];
  float* out = (float*)d_out;

  // workspace: nx f16 (8MB) | xh f16 (8MB) | Wh f16 (128KB) | value f32 (16MB)
  //          | colsum f32 (4KB) | partial f32 (256KB)
  char* ws = (char*)d_ws;
  const size_t nElems = (size_t)B_ * N_ * D_;
  _Float16* nx     = (_Float16*)ws;
  _Float16* xh     = (_Float16*)(ws + nElems * 2);
  _Float16* Wh     = (_Float16*)(ws + nElems * 4);
  float*    value  = (float*)   (ws + nElems * 4 + (size_t)D_ * D_ * 2);
  float*    colsum = (float*)   (ws + nElems * 4 + (size_t)D_ * D_ * 2 + nElems * 4);
  float*    partial= (float*)   (ws + nElems * 4 + (size_t)D_ * D_ * 2 + nElems * 4
                                    + (size_t)B_ * D_ * 4);

  prep_normalize<<<B_ * N_ / 8, 256, 0, stream>>>(x, nx, xh);
  convert_w     <<<(D_ * D_) / (256 * 4), 256, 0, stream>>>(W, Wh);
  value_gemm    <<<B_ * N_ / 16, 256, 0, stream>>>(xh, Wh, bias, value);
  colsum_partial<<<NSPLIT_ * 4, 256, 0, stream>>>(value, partial);
  colsum_reduce <<<(B_ * D_) / 256, 256, 0, stream>>>(partial, colsum);

  hipFuncSetAttribute((const void*)cos_topk_out,
                      hipFuncAttributeMaxDynamicSharedMemorySize, (int)SMEM_BYTES);
  cos_topk_out<<<B_ * N_ / 16, 512, SMEM_BYTES, stream>>>(nx, value, colsum, out);
}
